// quadratic_24515673325958
// MI455X (gfx1250) — matrix operand order repaired and checked by reading, not compiled, not run on hardware
//
#include <hip/hip_runtime.h>

typedef __attribute__((ext_vector_type(16))) _Float16 v16h;
typedef __attribute__((ext_vector_type(8)))  float    v8f;
typedef __attribute__((ext_vector_type(4)))  float    f32x4;
typedef __attribute__((ext_vector_type(4)))  _Float16 h4;

#define F_DIM    64
#define MBLK     128
#define NTHREADS 128

#define WMMA_F16(a, b, c) \
    __builtin_amdgcn_wmma_f32_16x16x32_f16(false, (a), false, (b), (short)0, (c), false, false)

__device__ __forceinline__ v16h cvt16(const float* p) {
    const f32x4* p4 = (const f32x4*)p;
    v16h r;
    #pragma unroll
    for (int q = 0; q < 4; ++q) {
        f32x4 t = p4[q];
        #pragma unroll
        for (int j = 0; j < 4; ++j) r[q * 4 + j] = (_Float16)t[j];
    }
    return r;
}

// ---- one-shot prep: W f32 -> f16 (row-major, same indexing) into d_ws ----
__global__ __launch_bounds__(256) void w_to_f16(const float* __restrict__ W,
                                                _Float16* __restrict__ Wh) {
    const int i = (blockIdx.x * 256 + threadIdx.x) * 4;
    f32x4 v = *(const f32x4*)(W + i);
    h4 h;
    h[0] = (_Float16)v[0]; h[1] = (_Float16)v[1];
    h[2] = (_Float16)v[2]; h[3] = (_Float16)v[3];
    *(h4*)(Wh + i) = h;
}

// out[b,o] = sum_{f,g} x[b,f]*x[b,g]*W[o,f*64+g]
// GEMM: A[b,k] = x[b,f]*x[b,g] built on the fly (f constant per 32-K chunk),
// B[k,o] = W[o,k].  Wave tile: 2 M-tiles x 4 N-tiles (32 rows x 64 cols):
// each A frag feeds 4 WMMAs (2 pk_mul/WMMA), B frags are shared loads
// across all 4 waves of the block (L0 broadcast).
template <bool WS16>
__global__ __launch_bounds__(NTHREADS) void quad_wmma_kernel(
    const float* __restrict__ x, const float* __restrict__ W,
    const _Float16* __restrict__ Wh, float* __restrict__ out)
{
    __shared__ float lds_x[MBLK * F_DIM];   // 32 KB

    const int t    = threadIdx.x;
    const int wave = t >> 5;           // 0..3 = M-group (32 rows each)
    const int lane = t & 31;
    const int lm   = lane & 15;
    const int hi   = lane >> 4;
    const long rowbase = (long)blockIdx.x * MBLK;

    // ---- stage 128 x-rows (8192 f32) into LDS ----
    {
        const f32x4* g4 = (const f32x4*)(x + rowbase * F_DIM);
        f32x4*       l4 = (f32x4*)lds_x;
        #pragma unroll
        for (int i = 0; i < 16; ++i) l4[t + NTHREADS * i] = g4[t + NTHREADS * i];
    }
    __syncthreads();

    // ---- per-lane x slices in A-frag layout (16-bit A 16x32, contiguous K order, same as B frags):
    // lane<16: row lm, K=0..15; lane>=16: row lm, K=16..31
    const int koff = hi ? 16 : 0;
    const int m0 = wave * 32 + lm;
    const int m1 = m0 + 16;
    v16h xlo0, xhi0, xlo1, xhi1;
    {
        const float* r0 = lds_x + m0 * F_DIM;
        const float* r1 = lds_x + m1 * F_DIM;
        #pragma unroll
        for (int e = 0; e < 16; ++e) {
            const int kk = koff + e;
            xlo0[e] = (_Float16)r0[kk];
            xhi0[e] = (_Float16)r0[32 + kk];
            xlo1[e] = (_Float16)r1[kk];
            xhi1[e] = (_Float16)r1[32 + kk];
        }
    }

    v8f c[2][4] = {};   // [mt][nt]: 32 rows x 64 cols

    // B-frag (32x16 f16): lane<16 holds K=0..15 of col n, lane>=16 K=16..31
    const int kkb = hi ? 16 : 0;
    const long nstride = (long)16 * (F_DIM * F_DIM);
    const float*    wf = W  + (long)lm * (F_DIM * F_DIM);
    const _Float16* wh = Wh + (long)lm * (F_DIM * F_DIM);

    for (int f = 0; f < F_DIM; ++f) {
        v16h b[4][2];   // [nt][g0-half]
        #pragma unroll
        for (int nt = 0; nt < 4; ++nt) {
            if constexpr (WS16) {
                const _Float16* p = wh + nt * nstride + f * F_DIM;
                b[nt][0] = *(const v16h*)(p + kkb);
                b[nt][1] = *(const v16h*)(p + 32 + kkb);
            } else {
                const float* p = wf + nt * nstride + f * F_DIM;
                b[nt][0] = cvt16(p + kkb);
                b[nt][1] = cvt16(p + 32 + kkb);
            }
        }
        const _Float16 s0 = (_Float16)lds_x[m0 * F_DIM + f];
        const _Float16 s1 = (_Float16)lds_x[m1 * F_DIM + f];

        // g0 = 0 pass: 8 WMMAs from 2 A frags
        v16h a0 = xlo0 * s0;
        v16h a1 = xlo1 * s1;
        #pragma unroll
        for (int nt = 0; nt < 4; ++nt) c[0][nt] = WMMA_F16(a0, b[nt][0], c[0][nt]);
        #pragma unroll
        for (int nt = 0; nt < 4; ++nt) c[1][nt] = WMMA_F16(a1, b[nt][0], c[1][nt]);
        // g0 = 32 pass (fresh A regs -> long WAR distances)
        v16h e0 = xhi0 * s0;
        v16h e1 = xhi1 * s1;
        #pragma unroll
        for (int nt = 0; nt < 4; ++nt) c[0][nt] = WMMA_F16(e0, b[nt][1], c[0][nt]);
        #pragma unroll
        for (int nt = 0; nt < 4; ++nt) c[1][nt] = WMMA_F16(e1, b[nt][1], c[1][nt]);
    }

    // ---- store: C/D layout — VGPR v: lanes0-15 -> M=v, lanes16-31 -> M=v+8
    #pragma unroll
    for (int mt = 0; mt < 2; ++mt) {
        const long rb = rowbase + wave * 32 + mt * 16 + hi * 8;
        float* o = out + rb * F_DIM + lm;
        #pragma unroll
        for (int v = 0; v < 8; ++v) {
            #pragma unroll
            for (int nt = 0; nt < 4; ++nt)
                o[(long)v * F_DIM + nt * 16] = c[mt][nt][v];
        }
    }
}

extern "C" void kernel_launch(void* const* d_in, const int* in_sizes, int n_in,
                              void* d_out, int out_size, void* d_ws, size_t ws_size,
                              hipStream_t stream) {
    const float* x = (const float*)d_in[0];   // [32768, 64] f32
    const float* W = (const float*)d_in[1];   // [64, 4096] f32
    float* out = (float*)d_out;               // [32768, 64] f32
    const int rows   = in_sizes[0] / F_DIM;   // 32768
    const int welems = in_sizes[1];           // 262144

    if (ws_size >= (size_t)welems * sizeof(_Float16)) {
        _Float16* Wh = (_Float16*)d_ws;
        w_to_f16<<<dim3(welems / (256 * 4)), dim3(256), 0, stream>>>(W, Wh);
        quad_wmma_kernel<true><<<dim3(rows / MBLK), dim3(NTHREADS), 0, stream>>>(
            x, W, Wh, out);
    } else {
        quad_wmma_kernel<false><<<dim3(rows / MBLK), dim3(NTHREADS), 0, stream>>>(
            x, W, nullptr, out);
    }
}
